// Decoder_9277129360123
// MI455X (gfx1250) — matrix operand order repaired and checked by reading, not compiled, not run on hardware
//
#include <hip/hip_runtime.h>

// ---------------------------------------------------------------- types
typedef __attribute__((ext_vector_type(16))) __bf16 v16bf;
typedef __attribute__((ext_vector_type(8)))  __bf16 bf16x8;
typedef __attribute__((ext_vector_type(8)))  float  v8f;
typedef __attribute__((ext_vector_type(4)))  int    i4v;

#define AS1 __attribute__((address_space(1)))
#define AS3 __attribute__((address_space(3)))

#if __has_builtin(__builtin_amdgcn_global_load_async_to_lds_b128)
#define HAVE_ASYNC_LDS 1
#else
#define HAVE_ASYNC_LDS 0
#endif

// ---------------------------------------------------------------- consts
constexpr int B_  = 2;
constexpr int S_  = 2048;
constexpr int E_  = 2048;
constexpr int H_  = 16;
constexpr int D_  = 128;            // head dim
constexpr int M_  = B_ * S_;        // 4096 rows for projections
constexpr int NQKV = 3 * E_;        // 6144

// ---------------------------------------------------------------- helpers
__device__ __forceinline__ __bf16 f2bf(float x) {
  unsigned u = __builtin_bit_cast(unsigned, x);
  unsigned r = (u + 0x7FFFu + ((u >> 16) & 1u)) >> 16;   // RNE
  unsigned short s = (unsigned short)r;
  return __builtin_bit_cast(__bf16, s);
}

// Build a v16bf fragment from two contiguous 8-element (16B) chunks.
__device__ __forceinline__ v16bf make_frag(const __bf16* p0, const __bf16* p1) {
  bf16x8 lo = *(const bf16x8*)p0;
  bf16x8 hi = *(const bf16x8*)p1;
  v16bf r;
#pragma unroll
  for (int i = 0; i < 8; ++i) { r[i] = lo[i]; r[i + 8] = hi[i]; }
  return r;
}

__device__ __forceinline__ void wait_async0() {
#if HAVE_ASYNC_LDS
#if __has_builtin(__builtin_amdgcn_s_wait_asynccnt)
  __builtin_amdgcn_s_wait_asynccnt(0);
#else
  asm volatile("s_wait_asynccnt 0x0" ::: "memory");
#endif
#endif
}

// ---------------------------------------------------------------- f32 -> bf16
__global__ __launch_bounds__(256) void cvt_bf16_kernel(
    const float* __restrict__ src, __bf16* __restrict__ dst, int n) {
  int i = (blockIdx.x * 256 + threadIdx.x) * 4;
  if (i >= n) return;
  float4 v = *(const float4*)(src + i);
  dst[i + 0] = f2bf(v.x);
  dst[i + 1] = f2bf(v.y);
  dst[i + 2] = f2bf(v.z);
  dst[i + 3] = f2bf(v.w);
}

// ---------------------------------------------------------------- GEMM
// C[M,N] = A[M,K] @ Bw[N,K]^T + bias
// Block 256 thr = 8 waves in a 4(m) x 2(n) grid; wave tile 64x64;
// block tile 256(m) x 128(n); K staged in LDS in steps of 32 via
// GLOBAL_LOAD_ASYNC_TO_LDS_B128 (ASYNCcnt) when available.
// MODE 0: epilogue scatters bf16 into q/[b][h][s][d], k/[b][h][s][d],
//         vT/[b][h][d][s].  MODE 1: f32 out += bias.
template <int MODE>
__global__ __launch_bounds__(256) void gemm_kernel(
    const __bf16* __restrict__ A, const __bf16* __restrict__ Bw,
    const float* __restrict__ bias,
    __bf16* __restrict__ qout, __bf16* __restrict__ kout,
    __bf16* __restrict__ vtout, float* __restrict__ fout) {
  constexpr int K = E_;
  __shared__ __align__(16) __bf16 As[256][32];
  __shared__ __align__(16) __bf16 Bs[128][32];

  const int tid = threadIdx.x;
  const int m0 = blockIdx.y * 256, n0 = blockIdx.x * 128;
  const int w = tid >> 5, L = tid & 31, half = L >> 4, l16 = L & 15;
  const int wm = w >> 1, wn = w & 1;          // 4x2 wave grid
  const int rB = tid >> 1, kB = (tid & 1) * 16;

  v8f acc[4][4];
#pragma unroll
  for (int i = 0; i < 4; ++i)
#pragma unroll
    for (int j = 0; j < 4; ++j) acc[i][j] = v8f{0, 0, 0, 0, 0, 0, 0, 0};

  const __bf16* gA = A + (size_t)(m0 + tid) * K;        // 32 elems / thread
  const __bf16* gB = Bw + (size_t)(n0 + rB) * K + kB;   // 16 elems / thread

  for (int k0 = 0; k0 < K; k0 += 32) {
#if HAVE_ASYNC_LDS
    {
      AS1 i4v* ga = (AS1 i4v*)(gA + k0);
      AS3 i4v* la = (AS3 i4v*)(&As[tid][0]);
      __builtin_amdgcn_global_load_async_to_lds_b128(ga, la, 0, 0);
      __builtin_amdgcn_global_load_async_to_lds_b128(ga, la, 16, 0);
      __builtin_amdgcn_global_load_async_to_lds_b128(ga, la, 32, 0);
      __builtin_amdgcn_global_load_async_to_lds_b128(ga, la, 48, 0);
      AS1 i4v* gb = (AS1 i4v*)(gB + k0);
      AS3 i4v* lb = (AS3 i4v*)(&Bs[rB][kB]);
      __builtin_amdgcn_global_load_async_to_lds_b128(gb, lb, 0, 0);
      __builtin_amdgcn_global_load_async_to_lds_b128(gb, lb, 16, 0);
    }
    __builtin_prefetch(gA + k0 + 32, 0, 3);
    __builtin_prefetch(gB + k0 + 32, 0, 3);
    wait_async0();
#else
    {
      const __bf16* pa = gA + k0;
#pragma unroll
      for (int c = 0; c < 4; ++c)
        *(uint4*)(&As[tid][c * 8]) = *(const uint4*)(pa + c * 8);
      const __bf16* pb = gB + k0;
      *(uint4*)(&Bs[rB][kB])     = *(const uint4*)(pb);
      *(uint4*)(&Bs[rB][kB + 8]) = *(const uint4*)(pb + 8);
    }
    __builtin_prefetch(gA + k0 + 32, 0, 3);
    __builtin_prefetch(gB + k0 + 32, 0, 3);
#endif
    __syncthreads();

    v16bf af[4];
#pragma unroll
    for (int i = 0; i < 4; ++i) {
      const int r = wm * 64 + i * 16 + l16;
      af[i] = make_frag(&As[r][8 * half], &As[r][16 + 8 * half]);
    }
#pragma unroll
    for (int j = 0; j < 4; ++j) {
      const int r = wn * 64 + j * 16 + l16;
      v16bf bfr = make_frag(&Bs[r][8 * half], &Bs[r][16 + 8 * half]);
#pragma unroll
      for (int i = 0; i < 4; ++i)
        acc[i][j] = __builtin_amdgcn_wmma_f32_16x16x32_bf16(
            false, af[i], false, bfr, (short)0, acc[i][j], false, false);
    }
    __syncthreads();
  }

#pragma unroll
  for (int i = 0; i < 4; ++i) {
#pragma unroll
    for (int j = 0; j < 4; ++j) {
      const int mb = m0 + wm * 64 + i * 16;
      const int n  = n0 + wn * 64 + j * 16 + l16;
      const float bv = bias[n];
      if constexpr (MODE == 0) {
        const int which = n >> 11;            // n / 2048
        const int h = (n >> 7) & (H_ - 1);
        const int d = n & (D_ - 1);
#pragma unroll
        for (int r = 0; r < 8; ++r) {
          const int m = mb + r + 8 * half;
          const int b = m >> 11;              // m / 2048
          const int s = m & (S_ - 1);
          const __bf16 o = f2bf(acc[i][j][r] + bv);
          const size_t bh = (size_t)(b * H_ + h);
          if (which == 0)      qout[(bh * S_ + s) * D_ + d] = o;
          else if (which == 1) kout[(bh * S_ + s) * D_ + d] = o;
          else                 vtout[(bh * D_ + d) * S_ + s] = o;  // V^T
        }
      } else {
#pragma unroll
        for (int r = 0; r < 8; ++r) {
          const int m = mb + r + 8 * half;
          fout[(size_t)m * E_ + n] = acc[i][j][r] + bv;
        }
      }
    }
  }
}

// ---------------------------------------------------------------- RoPE
// rotated[d] = cos(s*ang[d/2])*x[d] + sin(s*ang[d/2])*T[d]
// T[d] = -x[2d+1] (d<64) ; x[2(d-64)] (d>=64)   (exactly the reference)
__global__ __launch_bounds__(256) void rope_kernel(
    __bf16* __restrict__ qb, __bf16* __restrict__ kb) {
  const int t = blockIdx.x * 256 + threadIdx.x;   // 0 .. 2*B*H*S-1
  const int which = t >> 16;                      // B*H*S = 65536
  const int row = t & 65535;
  const int s = row & (S_ - 1);
  __bf16* p = (which ? kb : qb) + (size_t)row * D_;

  float x[D_];
#pragma unroll
  for (int d = 0; d < D_; ++d) x[d] = (float)p[d];
#pragma unroll
  for (int d = 0; d < D_; ++d) {
    const int j = d >> 1;
    // ang = s * 10000^(-(2j)/2048) ; ln(10000)/2048 = 4.4972365097539e-3
    const float ang = (float)s * __expf(-(float)(2 * j) * 4.4972365097539e-3f);
    float si, c;
    __sincosf(ang, &si, &c);
    const float tr = (d < 64) ? -x[2 * d + 1] : x[2 * (d - 64)];
    p[d] = f2bf(c * x[d] + si * tr);
  }
}

// ---------------------------------------------------------------- attention
// One wave handles 16 q-rows of one (b,h); flash-style over k in steps of 32.
__global__ __launch_bounds__(128) void attn_kernel(
    const __bf16* __restrict__ qb, const __bf16* __restrict__ kb,
    const __bf16* __restrict__ vt, const int* __restrict__ mask,
    __bf16* __restrict__ ctxout) {
  __shared__ __align__(16) float Ps[4][16][32];   // per-wave P staging

  const int bh = blockIdx.y;
  const int b = bh >> 4;                          // bh / H_
  const int w = threadIdx.x >> 5, L = threadIdx.x & 31;
  const int half = L >> 4, l16 = L & 15;
  const int q0 = blockIdx.x * 64 + w * 16;
  const size_t qkbase = (size_t)bh * S_ * D_;
  const size_t vbase  = (size_t)bh * D_ * S_;
  const float rs = 0.08838834764831845f;          // 1/sqrt(128)

  // Q fragments (A-layout): lane -> row q0 + l16, d-chunks of 32
  v16bf qf[4];
  {
    const __bf16* qp = qb + qkbase + (size_t)(q0 + l16) * D_;
#pragma unroll
    for (int f = 0; f < 4; ++f)
      qf[f] = make_frag(qp + f * 32 + 8 * half, qp + f * 32 + 16 + 8 * half);
  }

  v8f acc[8];
#pragma unroll
  for (int i = 0; i < 8; ++i) acc[i] = v8f{0, 0, 0, 0, 0, 0, 0, 0};
  float mrow[8], lrow[8];
#pragma unroll
  for (int r = 0; r < 8; ++r) { mrow[r] = -3.0e38f; lrow[r] = 0.f; }

#pragma unroll 1
  for (int k0 = 0; k0 < S_; k0 += 32) {
    // ---- scores: two 16x16 tiles covering k cols [k0, k0+32)
    v8f sc[2];
#pragma unroll
    for (int t = 0; t < 2; ++t) {
      const int krow = k0 + t * 16 + l16;
      const __bf16* kp = kb + qkbase + (size_t)krow * D_;
      v8f c = v8f{0, 0, 0, 0, 0, 0, 0, 0};
#pragma unroll
      for (int f = 0; f < 4; ++f) {
        v16bf kf = make_frag(kp + f * 32 + 8 * half, kp + f * 32 + 16 + 8 * half);
        c = __builtin_amdgcn_wmma_f32_16x16x32_bf16(
            false, qf[f], false, kf, (short)0, c, false, false);
      }
      const int mv = mask[b * S_ + krow];
#pragma unroll
      for (int r = 0; r < 8; ++r) c[r] = mv ? c[r] * rs : -1.0e9f;
      sc[t] = c;
    }

    // ---- online softmax (row reductions across the 16-lane half)
    float alpha[8];
#pragma unroll
    for (int r = 0; r < 8; ++r) {
      float rm = fmaxf(sc[0][r], sc[1][r]);
      rm = fmaxf(rm, __shfl_xor(rm, 1, 32));
      rm = fmaxf(rm, __shfl_xor(rm, 2, 32));
      rm = fmaxf(rm, __shfl_xor(rm, 4, 32));
      rm = fmaxf(rm, __shfl_xor(rm, 8, 32));
      const float mn = fmaxf(mrow[r], rm);
      alpha[r] = __expf(mrow[r] - mn);
      mrow[r] = mn;
      const float p0 = __expf(sc[0][r] - mn);
      const float p1 = __expf(sc[1][r] - mn);
      sc[0][r] = p0; sc[1][r] = p1;
      float rsum = p0 + p1;
      rsum += __shfl_xor(rsum, 1, 32);
      rsum += __shfl_xor(rsum, 2, 32);
      rsum += __shfl_xor(rsum, 4, 32);
      rsum += __shfl_xor(rsum, 8, 32);
      lrow[r] = lrow[r] * alpha[r] + rsum;
    }
#pragma unroll
    for (int dt = 0; dt < 8; ++dt)
#pragma unroll
      for (int r = 0; r < 8; ++r) acc[dt][r] *= alpha[r];

    // ---- C-layout P -> LDS -> A-layout bf16 fragment (wave-private region)
#pragma unroll
    for (int r = 0; r < 8; ++r) {
      Ps[w][r + 8 * half][l16]      = sc[0][r];
      Ps[w][r + 8 * half][16 + l16] = sc[1][r];
    }
    __builtin_amdgcn_wave_barrier();
    asm volatile("s_wait_dscnt 0x0" ::: "memory");
    v16bf pf;
#pragma unroll
    for (int i = 0; i < 8; ++i) {
      pf[i]     = f2bf(Ps[w][l16][8 * half + i]);
      pf[i + 8] = f2bf(Ps[w][l16][16 + 8 * half + i]);
    }
    __builtin_amdgcn_wave_barrier();

    // ---- ctx += P @ V   (V stored transposed: [b][h][d][s])
#pragma unroll
    for (int dt = 0; dt < 8; ++dt) {
      const __bf16* vp = vt + vbase + (size_t)(dt * 16 + l16) * S_ + k0;
      v16bf vf = make_frag(vp + 8 * half, vp + 16 + 8 * half);
      acc[dt] = __builtin_amdgcn_wmma_f32_16x16x32_bf16(
          false, pf, false, vf, (short)0, acc[dt], false, false);
    }
  }

  // ---- normalize and store ctx as [b][s][h*128 + d] bf16
#pragma unroll
  for (int dt = 0; dt < 8; ++dt) {
#pragma unroll
    for (int r = 0; r < 8; ++r) {
      const int q = q0 + r + 8 * half;
      const int e = (bh & (H_ - 1)) * D_ + dt * 16 + l16;
      ctxout[(size_t)(b * S_ + q) * E_ + e] = f2bf(acc[dt][r] / lrow[r]);
    }
  }
}

// ---------------------------------------------------------------- launch
extern "C" void kernel_launch(void* const* d_in, const int* in_sizes, int n_in,
                              void* d_out, int out_size, void* d_ws, size_t ws_size,
                              hipStream_t stream) {
  const float* X    = (const float*)d_in[0];
  const int*   mask = (const int*)d_in[1];
  const float* Wqkv = (const float*)d_in[2];
  const float* bqkv = (const float*)d_in[3];
  const float* Wo   = (const float*)d_in[4];
  const float* bo   = (const float*)d_in[5];
  float* out = (float*)d_out;

  // workspace layout (bf16 elements)
  constexpr size_t XBF_N  = (size_t)M_ * E_;      // 8,388,608
  constexpr size_t WQKV_N = (size_t)NQKV * E_;    // 12,582,912
  constexpr size_t WO_N   = (size_t)E_ * E_;      // 4,194,304
  __bf16* xbf   = (__bf16*)d_ws;
  __bf16* wqkvb = xbf + XBF_N;
  __bf16* wob   = wqkvb + WQKV_N;
  __bf16* qbf   = wob + WO_N;
  __bf16* kbf   = qbf + XBF_N;
  __bf16* vtb   = kbf + XBF_N;
  __bf16* ctxb  = vtb + XBF_N;

  // 1) convert inputs to bf16
  cvt_bf16_kernel<<<(int)(XBF_N / 1024), 256, 0, stream>>>(X, xbf, (int)XBF_N);
  cvt_bf16_kernel<<<(int)(WQKV_N / 1024), 256, 0, stream>>>(Wqkv, wqkvb, (int)WQKV_N);
  cvt_bf16_kernel<<<(int)(WO_N / 1024), 256, 0, stream>>>(Wo, wob, (int)WO_N);

  // 2) QKV projection (+ scatter into q/k/vT layouts)
  gemm_kernel<0><<<dim3(NQKV / 128, M_ / 256), 256, 0, stream>>>(
      xbf, wqkvb, bqkv, qbf, kbf, vtb, nullptr);

  // 3) RoPE on Q and K (in place)
  rope_kernel<<<(2 * B_ * H_ * S_) / 256, 256, 0, stream>>>(qbf, kbf);

  // 4) flash attention
  attn_kernel<<<dim3(S_ / 64, B_ * H_), 128, 0, stream>>>(
      qbf, kbf, vtb, mask, ctxb);

  // 5) output projection
  gemm_kernel<1><<<dim3(E_ / 128, M_ / 256), 256, 0, stream>>>(
      ctxb, wob, bo, nullptr, nullptr, nullptr, out);
}
